// Head_27479200760434
// MI455X (gfx1250) — matrix operand order repaired and checked by reading, not compiled, not run on hardware
//
#include <hip/hip_runtime.h>

typedef __attribute__((ext_vector_type(16))) __bf16 v16bf;
typedef __attribute__((ext_vector_type(8)))  __bf16 v8bf;
typedef __attribute__((ext_vector_type(8)))  float  v8f;

constexpr int Bn = 256, Tn = 256, Cn = 384, HSn = 64;
constexpr int NTOK = Bn * Tn * HSn;   // elements per q/k/v tensor
constexpr int WTN  = Cn * HSn;        // 24576 per weight matrix

// padded LDS strides (elements) -> conflict-free 16-lane groups (stride/2 dwords, gcd(.,64)=4)
constexpr int CP = Cn + 8;            // 392: weight row stride in LDS
constexpr int MP = HSn * CP;          // 25088: one weight matrix in LDS
constexpr int DP = HSn + 8;           // 72:  K row stride in LDS
constexpr int TP = Tn + 8;            // 264: V^T row stride in LDS

constexpr int QKV_LDS_BYTES  = 3 * MP * 2;                        // 150528
constexpr int ATTN_LDS_BYTES = (Tn * DP + HSn * TP + 8 * 512) * 2; // 78848

#define WMMA_BF16(a, b, c) \
  __builtin_amdgcn_wmma_f32_16x16x32_bf16(false, (a), false, (b), (short)0, (c), false, false)

__device__ __forceinline__ unsigned short f2bf(float f) {
  __bf16 h = (__bf16)f;
  return __builtin_bit_cast(unsigned short, h);
}

// ---------------- kernel 0: weights -> bf16, transposed [n][c] ----------------
__global__ void wt_kernel(const float* __restrict__ Wq, const float* __restrict__ Wk,
                          const float* __restrict__ Wv, unsigned short* __restrict__ WT) {
  int idx = blockIdx.x * blockDim.x + threadIdx.x;
  if (idx >= 3 * WTN) return;
  int m = idx / WTN, r = idx % WTN;
  int n = r / Cn, c = r % Cn;
  const float* W = (m == 0) ? Wq : ((m == 1) ? Wk : Wv);
  WT[idx] = f2bf(W[c * HSn + n]);
}

// ---------------- kernel 1: QKV projection, weights staged in LDS -------------
__global__ __launch_bounds__(256) void qkv_kernel(
    const float* __restrict__ x, const unsigned short* __restrict__ WTu,
    const float* __restrict__ bq, const float* __restrict__ bk, const float* __restrict__ bv,
    unsigned short* __restrict__ qo, unsigned short* __restrict__ ko,
    unsigned short* __restrict__ vTo) {
  extern __shared__ char smem[];
  __bf16* wlds = (__bf16*)smem;                 // [3][64][CP]

  const int lane = threadIdx.x & 31;
  const int wave = threadIdx.x >> 5;
  const int task = blockIdx.x * 8 + wave;       // 0..4095
  const int b    = task >> 4;
  const int t0   = (task & 15) << 4;
  const int l16  = lane & 15;
  const int g    = lane >> 4;
  const __bf16* WT = (const __bf16*)WTu;

  // ---- cooperative stage: 3*64*384 bf16 -> padded LDS (36 x 16B per thread)
#pragma unroll
  for (int i = 0; i < 36; ++i) {
    const int cid = i * 256 + threadIdx.x;      // 0..9215
    const int m  = cid / 3072;
    const int r  = cid % 3072;
    const int n  = r / 48;
    const int c8 = (r % 48) * 8;
    *(v8bf*)(wlds + m * MP + n * CP + c8) =
        *(const v8bf*)(WT + m * WTN + n * Cn + c8);
  }
  __syncthreads();

  v8f acc[3][4];
#pragma unroll
  for (int m = 0; m < 3; ++m)
#pragma unroll
    for (int nt = 0; nt < 4; ++nt) acc[m][nt] = (v8f){};

  const float* xrow = x + (size_t)(b * Tn + t0 + l16) * Cn;
  const __bf16* wlane = wlds + (size_t)l16 * CP + g * 16;

#pragma unroll
  for (int kk = 0; kk < 12; ++kk) {
    const int k0 = kk * 32 + g * 16;
    float4 f0 = *(const float4*)(xrow + k0);
    float4 f1 = *(const float4*)(xrow + k0 + 4);
    float4 f2 = *(const float4*)(xrow + k0 + 8);
    float4 f3 = *(const float4*)(xrow + k0 + 12);
    v16bf a;
    a[0]  = (__bf16)f0.x; a[1]  = (__bf16)f0.y; a[2]  = (__bf16)f0.z; a[3]  = (__bf16)f0.w;
    a[4]  = (__bf16)f1.x; a[5]  = (__bf16)f1.y; a[6]  = (__bf16)f1.z; a[7]  = (__bf16)f1.w;
    a[8]  = (__bf16)f2.x; a[9]  = (__bf16)f2.y; a[10] = (__bf16)f2.z; a[11] = (__bf16)f2.w;
    a[12] = (__bf16)f3.x; a[13] = (__bf16)f3.y; a[14] = (__bf16)f3.z; a[15] = (__bf16)f3.w;
#pragma unroll
    for (int m = 0; m < 3; ++m) {
#pragma unroll
      for (int nt = 0; nt < 4; ++nt) {
        const __bf16* wp = wlane + m * MP + nt * 16 * CP + kk * 32;
        v16bf bm = *(const v16bf*)wp;           // 2x ds_load_b128, bank-clean
        acc[m][nt] = WMMA_BF16(a, bm, acc[m][nt]);
      }
    }
  }

#pragma unroll
  for (int nt = 0; nt < 4; ++nt) {
    const int n = nt * 16 + l16;
    const float bqv = bq[n], bkv = bk[n], bvv = bv[n];
    v8bf pk;
#pragma unroll
    for (int r = 0; r < 8; ++r) {
      const int t = t0 + r + 8 * g;
      const size_t o = (size_t)(b * Tn + t) * HSn + n;
      qo[o] = f2bf(acc[0][nt][r] + bqv);
      ko[o] = f2bf(acc[1][nt][r] + bkv);
      pk[r] = (__bf16)(acc[2][nt][r] + bvv);
    }
    *(v8bf*)(vTo + (size_t)(b * HSn + n) * Tn + t0 + 8 * g) = pk;
  }
}

// ---------------- kernel 2: causal flash attention, K/V staged in LDS ---------
__global__ __launch_bounds__(256) void attn_kernel(
    const unsigned short* __restrict__ qu, const unsigned short* __restrict__ ku,
    const unsigned short* __restrict__ vTu, float* __restrict__ out) {
  extern __shared__ char smem[];
  __bf16* klds = (__bf16*)smem;        // [256][DP]
  __bf16* vlds = klds + Tn * DP;       // [64][TP]
  __bf16* plds = vlds + HSn * TP;      // 8 waves x 512

  const int lane = threadIdx.x & 31;
  const int wave = threadIdx.x >> 5;
  const int task = blockIdx.x * 8 + wave;
  const int b    = task >> 4;          // same for all waves in the block
  const int qt0  = (task & 15) << 4;
  const int l16  = lane & 15;
  const int g    = lane >> 4;
  const __bf16* q  = (const __bf16*)qu;
  const __bf16* k  = (const __bf16*)ku;
  const __bf16* vT = (const __bf16*)vTu;

  // ---- stage K and V^T for this batch (only the causally needed rows)
  const int nrows = (blockIdx.x & 1) ? 256 : 128;
  for (int cid = threadIdx.x; cid < nrows * 8; cid += 256) {     // K: nrows x 64
    const int s = cid >> 3, c8 = (cid & 7) * 8;
    *(v8bf*)(klds + s * DP + c8) = *(const v8bf*)(k + (size_t)(b * Tn + s) * HSn + c8);
  }
  for (int cid = threadIdx.x; cid < 64 * 32; cid += 256) {       // V^T: 64 x 256
    const int d = cid >> 5, t8 = (cid & 31) * 8;
    if (t8 < nrows)
      *(v8bf*)(vlds + d * TP + t8) = *(const v8bf*)(vT + (size_t)(b * HSn + d) * Tn + t8);
  }
  __syncthreads();

  // ---- Q tile in A layout
  v16bf aq[2];
#pragma unroll
  for (int d = 0; d < 2; ++d) {
    const __bf16* qp = q + (size_t)(b * Tn + qt0 + l16) * HSn + d * 32 + g * 16;
    v8bf lo = *(const v8bf*)qp;
    v8bf hi = *(const v8bf*)(qp + 8);
    aq[d] = __builtin_shufflevector(lo, hi, 0, 1, 2, 3, 4, 5, 6, 7,
                                            8, 9, 10, 11, 12, 13, 14, 15);
  }

  v8f oacc[4];
#pragma unroll
  for (int nt = 0; nt < 4; ++nt) oacc[nt] = (v8f){};
  float m_[8], l_[8];
#pragma unroll
  for (int r = 0; r < 8; ++r) { m_[r] = -3.0e38f; l_[r] = 0.0f; }

  const float scale = 0.125f;   // 1/sqrt(64)
  const int jmax = qt0 >> 5;
  __bf16* pw = plds + wave * 512;

  for (int j = 0; j <= jmax; ++j) {
    const int st0 = j * 32;
    // strip operands from LDS (bank-clean padded strides)
    v16bf kb[2][2];
#pragma unroll
    for (int t2 = 0; t2 < 2; ++t2)
#pragma unroll
      for (int d = 0; d < 2; ++d)
        kb[t2][d] = *(const v16bf*)(klds + (st0 + t2 * 16 + l16) * DP + d * 32 + g * 16);
    v16bf vb[4];
#pragma unroll
    for (int nt = 0; nt < 4; ++nt)
      vb[nt] = *(const v16bf*)(vlds + (nt * 16 + l16) * TP + st0 + g * 16);

    // S = Q K^T
    v8f s0 = (v8f){}, s1 = (v8f){};
#pragma unroll
    for (int d = 0; d < 2; ++d) {
      s0 = WMMA_BF16(aq[d], kb[0][d], s0);
      s1 = WMMA_BF16(aq[d], kb[1][d], s1);
    }

    // online softmax over rows M = r + 8*g
#pragma unroll
    for (int r = 0; r < 8; ++r) {
      const int qg = qt0 + r + 8 * g;
      const bool u0 = (st0 + l16) <= qg;
      const bool u1 = (st0 + 16 + l16) <= qg;
      float v0 = u0 ? s0[r] * scale : -3.0e38f;
      float v1 = u1 ? s1[r] * scale : -3.0e38f;
      float cm = fmaxf(v0, v1);
#pragma unroll
      for (int off = 1; off < 16; off <<= 1) cm = fmaxf(cm, __shfl_xor(cm, off, 32));
      const float mn = fmaxf(m_[r], cm);
      const float al = __expf(m_[r] - mn);
      const float p0 = u0 ? __expf(v0 - mn) : 0.0f;
      const float p1 = u1 ? __expf(v1 - mn) : 0.0f;
      float rs = p0 + p1;
#pragma unroll
      for (int off = 1; off < 16; off <<= 1) rs += __shfl_xor(rs, off, 32);
      l_[r] = l_[r] * al + rs;
      m_[r] = mn;
#pragma unroll
      for (int nt = 0; nt < 4; ++nt) oacc[nt][r] *= al;
      pw[(r + 8 * g) * 32 + l16]      = (__bf16)p0;
      pw[(r + 8 * g) * 32 + 16 + l16] = (__bf16)p1;
    }

    // P (C-layout) -> A-layout via wave-private LDS bounce
    const __bf16* pp = pw + l16 * 32 + g * 16;
    v8bf lo = *(const v8bf*)pp;
    v8bf hi = *(const v8bf*)(pp + 8);
    v16bf ap = __builtin_shufflevector(lo, hi, 0, 1, 2, 3, 4, 5, 6, 7,
                                               8, 9, 10, 11, 12, 13, 14, 15);

    // O += P @ V
#pragma unroll
    for (int nt = 0; nt < 4; ++nt)
      oacc[nt] = WMMA_BF16(ap, vb[nt], oacc[nt]);
  }

#pragma unroll
  for (int nt = 0; nt < 4; ++nt) {
#pragma unroll
    for (int r = 0; r < 8; ++r) {
      const int t = qt0 + r + 8 * g;
      out[(size_t)(b * Tn + t) * HSn + nt * 16 + l16] = oacc[nt][r] / l_[r];
    }
  }
}

// ---------------- launcher ----------------------------------------------------
extern "C" void kernel_launch(void* const* d_in, const int* in_sizes, int n_in,
                              void* d_out, int out_size, void* d_ws, size_t ws_size,
                              hipStream_t stream) {
  const float* x  = (const float*)d_in[0];
  const float* Wq = (const float*)d_in[1];
  const float* bq = (const float*)d_in[2];
  const float* Wk = (const float*)d_in[3];
  const float* bk = (const float*)d_in[4];
  const float* Wv = (const float*)d_in[5];
  const float* bv = (const float*)d_in[6];

  unsigned short* qb = (unsigned short*)d_ws;
  unsigned short* kb = qb + (size_t)NTOK;
  unsigned short* vT = kb + (size_t)NTOK;
  unsigned short* WT = vT + (size_t)NTOK;

  wt_kernel<<<(3 * WTN + 255) / 256, 256, 0, stream>>>(Wq, Wk, Wv, WT);
  qkv_kernel<<<512, 256, QKV_LDS_BYTES, stream>>>(x, WT, bq, bk, bv, qb, kb, vT);
  attn_kernel<<<512, 256, ATTN_LDS_BYTES, stream>>>(qb, kb, vT, (float*)d_out);
}
